// Transformer_6717328851555
// MI455X (gfx1250) — matrix operand order repaired and checked by reading, not compiled, not run on hardware
//
#include <hip/hip_runtime.h>
#include <cmath>

typedef __attribute__((ext_vector_type(16))) __bf16 v16bf;
typedef __attribute__((ext_vector_type(8)))  float  v8f;

#define DEVINL __device__ __forceinline__

constexpr int kB = 4, kN = 2048, kD = 768, kH = 12, kDH = 64, kM = 3072, kL = 4;
constexpr int kT   = kB * kN;        // 8192 tokens
constexpr int kQKV = 3 * kH * kDH;   // 2304

union FragU { uint4 u[2]; v16bf v; };

// A-matrix 16x32 bf16 fragment from row-major LDS tile.
// Lanes 0-15: M=lane, K = {0..7, 16..23}; lanes 16-31: M=lane-16, K = {8..15, 24..31}.
DEVINL v16bf lds_frag_a(const __bf16* base, int row0, int col0, int stride, int lane) {
  const int m   = lane & 15;
  const int klo = (lane & 16) ? 8 : 0;
  const __bf16* p = base + (row0 + m) * stride + col0 + klo;
  FragU f;
  f.u[0] = *(const uint4*)(p);
  f.u[1] = *(const uint4*)(p + 16);
  return f.v;
}

// B-matrix 32x16 bf16 fragment from K-transposed LDS tile Bt[n][k].
// Lanes 0-15: N=lane, K = {0..7, 16..23}; lanes 16-31: N=lane-16, K = {8..15, 24..31}.
DEVINL v16bf lds_frag_b(const __bf16* base, int n0, int k0, int stride, int lane) {
  const int n  = lane & 15;
  const int ks = (lane & 16) ? 8 : 0;
  const __bf16* p = base + (n0 + n) * stride + k0 + ks;
  FragU f;
  f.u[0] = *(const uint4*)(p);
  f.u[1] = *(const uint4*)(p + 16);
  return f.v;
}

DEVINL v8f wmma_bf16(v16bf a, v16bf b, v8f c) {
  return __builtin_amdgcn_wmma_f32_16x16x32_bf16(false, a, false, b, (short)0, c, false, false);
}

// ---------------- weight transpose + f32 -> bf16 convert -------------------
// in: f32 [K][Nc] row-major ; out: bf16 [Nc][K] row-major
__global__ __launch_bounds__(256) void transpose_cvt_kernel(const float* __restrict__ in,
                                                            __bf16* __restrict__ out,
                                                            int K, int Nc) {
  __shared__ float tile[32][33];
  const int bx = blockIdx.x * 32, by = blockIdx.y * 32;
  const int tx = threadIdx.x, ty = threadIdx.y;
#pragma unroll
  for (int i = 0; i < 4; i++)
    tile[ty + i * 8][tx] = in[(size_t)(by + ty + i * 8) * Nc + bx + tx];
  __syncthreads();
#pragma unroll
  for (int i = 0; i < 4; i++)
    out[(size_t)(bx + ty + i * 8) * K + by + tx] = (__bf16)tile[tx][ty + i * 8];
}

// ---------------- layernorm ------------------------------------------------
template <bool F32OUT>
__global__ __launch_bounds__(256) void ln_kernel(const float* __restrict__ x,
                                                 const float* __restrict__ w,
                                                 const float* __restrict__ b,
                                                 __bf16* __restrict__ outB,
                                                 float* __restrict__ outF) {
  const int row = blockIdx.x, t = threadIdx.x;
  const float* xr = x + (size_t)row * kD;
  float s = 0.f, q = 0.f;
#pragma unroll
  for (int i = t; i < kD; i += 256) { float v = xr[i]; s += v; q += v * v; }
  __shared__ float rs[256], rq[256];
  rs[t] = s; rq[t] = q;
  __syncthreads();
  for (int o = 128; o > 0; o >>= 1) {
    if (t < o) { rs[t] += rs[t + o]; rq[t] += rq[t + o]; }
    __syncthreads();
  }
  const float mu   = rs[0] * (1.f / kD);
  const float var  = rq[0] * (1.f / kD) - mu * mu;
  const float rstd = rsqrtf(var + 1e-5f);
#pragma unroll
  for (int i = t; i < kD; i += 256) {
    float v = (xr[i] - mu) * rstd * w[i] + b[i];
    if (F32OUT) outF[(size_t)row * kD + i] = v;
    else        outB[(size_t)row * kD + i] = (__bf16)v;
  }
}

// ---------------- WMMA GEMM: C[M,Ncols] = A[M,K] * Bt[Ncols,K]^T -----------
// Block tile 128(M) x 256(N), BK=32, double-buffered LDS, 1 barrier / K-step.
// 8 waves in 2x4 grid, 64x64 per wave (16 WMMAs per K-step, 4x frag reuse).
// EPI: 0 = plain -> bf16 out
//      1 = +residual -> f32 out
//      2 = +bias, GELU -> bf16 out
//      3 = +bias, +residual -> f32 out
template <int EPI>
__global__ __launch_bounds__(256) void gemm_kernel(const __bf16* __restrict__ A,
                                                   const __bf16* __restrict__ Bt,
                                                   const float* __restrict__ bias,
                                                   const float* __restrict__ residual,
                                                   float* __restrict__ outF,
                                                   __bf16* __restrict__ outB,
                                                   int Ncols, int K) {
  constexpr bool HAS_BIAS = (EPI == 2 || EPI == 3);
  constexpr bool GELU     = (EPI == 2);
  constexpr bool RES      = (EPI == 1 || EPI == 3);
  constexpr bool F32OUT   = RES;
  __shared__ __attribute__((aligned(16))) __bf16 As[2][128 * 32];
  __shared__ __attribute__((aligned(16))) __bf16 Bs[2][256 * 32];
  const int t = threadIdx.x, lane = t & 31, wid = t >> 5;
  const int wm = wid >> 2, wn = wid & 3;             // 2 x 4 wave grid
  const int rb = blockIdx.y * 128, cb = blockIdx.x * 256;
  const int rowS = t >> 1, colCh = (t & 1) * 16;     // staging assignment
  const __bf16* gaBase  = A  + (size_t)(rb + rowS) * K + colCh;
  const __bf16* gbBase0 = Bt + (size_t)(cb + rowS) * K + colCh;
  const __bf16* gbBase1 = Bt + (size_t)(cb + rowS + 128) * K + colCh;

  uint4 ra0, ra1, rb0a, rb0b, rb1a, rb1b;            // register staging
  auto load_tile = [&](int kk) {
    ra0  = *(const uint4*)(gaBase + kk);
    ra1  = *(const uint4*)(gaBase + kk + 8);
    rb0a = *(const uint4*)(gbBase0 + kk);
    rb0b = *(const uint4*)(gbBase0 + kk + 8);
    rb1a = *(const uint4*)(gbBase1 + kk);
    rb1b = *(const uint4*)(gbBase1 + kk + 8);
  };
  auto store_tile = [&](int buf) {
    *(uint4*)(As[buf] + rowS * 32 + colCh)             = ra0;
    *(uint4*)(As[buf] + rowS * 32 + colCh + 8)         = ra1;
    *(uint4*)(Bs[buf] + rowS * 32 + colCh)             = rb0a;
    *(uint4*)(Bs[buf] + rowS * 32 + colCh + 8)         = rb0b;
    *(uint4*)(Bs[buf] + (rowS + 128) * 32 + colCh)     = rb1a;
    *(uint4*)(Bs[buf] + (rowS + 128) * 32 + colCh + 8) = rb1b;
  };

  v8f acc[4][4] = {};
  load_tile(0);
  store_tile(0);
  int buf = 0;
  for (int kk = 0; kk < K; kk += 32, buf ^= 1) {
    __syncthreads();                       // lds[buf] ready; lds[buf^1] free
    const bool more = (kk + 32 < K);
    if (more) {
      load_tile(kk + 32);                  // global loads overlap compute below
      if (kk + 64 < K) {
        __builtin_prefetch(gaBase + kk + 64, 0, 1);
        __builtin_prefetch(gbBase0 + kk + 64, 0, 1);
        __builtin_prefetch(gbBase1 + kk + 64, 0, 1);
      }
    }
    v16bf a[4];
#pragma unroll
    for (int i = 0; i < 4; i++)
      a[i] = lds_frag_a(As[buf], wm * 64 + i * 16, 0, 32, lane);
#pragma unroll
    for (int j = 0; j < 4; j++) {
      v16bf bj = lds_frag_b(Bs[buf], wn * 64 + j * 16, 0, 32, lane);
#pragma unroll
      for (int i = 0; i < 4; i++)
        acc[i][j] = wmma_bf16(a[i], bj, acc[i][j]);
    }
    if (more) store_tile(buf ^ 1);
  }
  // Epilogue. C layout: row = v + 8*(lane>=16), col = lane&15.
  const int rlo = (lane & 16) ? 8 : 0;
  const int c0  = lane & 15;
#pragma unroll
  for (int i = 0; i < 4; i++)
#pragma unroll
    for (int j = 0; j < 4; j++) {
      const int Cc = cb + wn * 64 + j * 16 + c0;
#pragma unroll
      for (int v = 0; v < 8; v++) {
        const size_t R = (size_t)(rb + wm * 64 + i * 16 + rlo + v);
        float val = acc[i][j][v];
        if (HAS_BIAS) val += bias[Cc];
        if (GELU)     val = 0.5f * val * (1.f + erff(val * 0.70710678118f));
        const size_t idx = R * (size_t)Ncols + Cc;
        if (RES) val += residual[idx];
        if (F32OUT) outF[idx] = val;
        else        outB[idx] = (__bf16)val;
      }
    }
}

// ---------------- flash attention ------------------------------------------
// grid: (N/128, B*H). Each block: 128 query rows for one (b,h); 8 waves x 16 rows.
__global__ __launch_bounds__(256) void attn_kernel(const __bf16* __restrict__ qkv,
                                                   __bf16* __restrict__ out) {
  __shared__ __attribute__((aligned(16))) __bf16 Qs[128 * 64];
  __shared__ __attribute__((aligned(16))) __bf16 Ks[64 * 64];   // [key][dh]
  __shared__ __attribute__((aligned(16))) __bf16 Vt[64 * 64];   // [dh][key]
  __shared__ __attribute__((aligned(16))) __bf16 Ps[8 * 16 * 64]; // per-wave P
  const int t = threadIdx.x, lane = t & 31, wid = t >> 5;
  const int qt = blockIdx.x, bh = blockIdx.y;
  const int b = bh / kH, h = bh % kH;
  const size_t rowbase = (size_t)b * kN + (size_t)qt * 128;
  const __bf16* Qg = qkv + rowbase * kQKV + h * kDH;
  const __bf16* Kg = qkv + (size_t)b * kN * kQKV + kD + h * kDH;
  const __bf16* Vg = Kg + kD;
  // stage Q tile (128 x 64)
#pragma unroll
  for (int i = 0; i < 4; i++) {
    const int e = t + i * 256, r = e >> 3, c = (e & 7) * 8;
    *(uint4*)(Qs + r * 64 + c) = *(const uint4*)(Qg + (size_t)r * kQKV + c);
  }
  float m[8], l[8];
#pragma unroll
  for (int v = 0; v < 8; v++) { m[v] = -3.0e38f; l[v] = 0.f; }
  v8f o[4] = {};
  const int rlo = (lane & 16) ? 8 : 0;
  const int c0  = lane & 15;

  for (int kt = 0; kt < kN / 64; kt++) {
    __syncthreads();
    const __bf16* kg = Kg + (size_t)kt * 64 * kQKV;
    const __bf16* vg = Vg + (size_t)kt * 64 * kQKV;
#pragma unroll
    for (int i = 0; i < 2; i++) {
      const int e = t + i * 256, r = e >> 3, c = (e & 7) * 8;
      *(uint4*)(Ks + r * 64 + c) = *(const uint4*)(kg + (size_t)r * kQKV + c);
      uint4 vv = *(const uint4*)(vg + (size_t)r * kQKV + c);
      const __bf16* vb = (const __bf16*)&vv;
#pragma unroll
      for (int j = 0; j < 8; j++) Vt[(c + j) * 64 + r] = vb[j];
    }
    __syncthreads();
    // S = Q * K^T  (n = key index, k = dh; Ks row-major is exactly Bt layout)
    v8f s[4] = {};
#pragma unroll
    for (int ks = 0; ks < 64; ks += 32) {
      v16bf a = lds_frag_a(Qs, wid * 16, ks, 64, lane);
#pragma unroll
      for (int j = 0; j < 4; j++) {
        v16bf bf = lds_frag_b(Ks, j * 16, ks, 64, lane);
        s[j] = wmma_bf16(a, bf, s[j]);
      }
    }
#pragma unroll
    for (int j = 0; j < 4; j++)
#pragma unroll
      for (int v = 0; v < 8; v++) s[j][v] *= 0.125f;   // DH^-0.5
    // online softmax; row stats reduced across each 16-lane half
    __bf16* pw = Ps + wid * 16 * 64;
#pragma unroll
    for (int v = 0; v < 8; v++) {
      float mx = -3.0e38f;
#pragma unroll
      for (int j = 0; j < 4; j++) mx = fmaxf(mx, s[j][v]);
      mx = fmaxf(mx, __shfl_xor(mx, 1, 32));
      mx = fmaxf(mx, __shfl_xor(mx, 2, 32));
      mx = fmaxf(mx, __shfl_xor(mx, 4, 32));
      mx = fmaxf(mx, __shfl_xor(mx, 8, 32));
      const float mn = fmaxf(m[v], mx);
      const float alpha = __expf(m[v] - mn);
      m[v] = mn;
      float ps = 0.f;
#pragma unroll
      for (int j = 0; j < 4; j++) {
        float p = __expf(s[j][v] - mn);
        s[j][v] = p;
        ps += p;
      }
      ps += __shfl_xor(ps, 1, 32);
      ps += __shfl_xor(ps, 2, 32);
      ps += __shfl_xor(ps, 4, 32);
      ps += __shfl_xor(ps, 8, 32);
      l[v] = l[v] * alpha + ps;
#pragma unroll
      for (int d = 0; d < 4; d++) o[d][v] *= alpha;
      const int prow = v + rlo;
#pragma unroll
      for (int j = 0; j < 4; j++) pw[prow * 64 + j * 16 + c0] = (__bf16)s[j][v];
    }
    // O += P * V  (n = dh, k = key; Vt is the needed Bt layout)
#pragma unroll
    for (int ks = 0; ks < 64; ks += 32) {
      v16bf a = lds_frag_a(pw, 0, ks, 64, lane);
#pragma unroll
      for (int d = 0; d < 4; d++) {
        v16bf bf = lds_frag_b(Vt, d * 16, ks, 64, lane);
        o[d] = wmma_bf16(a, bf, o[d]);
      }
    }
  }
  // normalize and write
#pragma unroll
  for (int d = 0; d < 4; d++)
#pragma unroll
    for (int v = 0; v < 8; v++) {
      const float val = o[d][v] / l[v];
      const size_t R = rowbase + wid * 16 + v + rlo;
      out[R * kD + h * kDH + d * 16 + c0] = (__bf16)val;
    }
}

// ---------------- host side ------------------------------------------------
extern "C" void kernel_launch(void* const* d_in, const int* in_sizes, int n_in,
                              void* d_out, int out_size, void* d_ws, size_t ws_size,
                              hipStream_t stream) {
  (void)in_sizes; (void)n_in; (void)out_size; (void)ws_size;
  const float* x_in  = (const float*)d_in[0];
  const float* ln1_w = (const float*)d_in[1];
  const float* ln1_b = (const float*)d_in[2];
  const float* w_qkv = (const float*)d_in[3];
  const float* w_out = (const float*)d_in[4];
  const float* ln2_w = (const float*)d_in[5];
  const float* ln2_b = (const float*)d_in[6];
  const float* w1    = (const float*)d_in[7];
  const float* b1    = (const float*)d_in[8];
  const float* w2    = (const float*)d_in[9];
  const float* b2    = (const float*)d_in[10];
  const float* fn_w  = (const float*)d_in[11];
  const float* fn_b  = (const float*)d_in[12];

  char* ws = (char*)d_ws;
  float*  xbuf    = (float*)ws;  ws += (size_t)kT * kD * sizeof(float);
  __bf16* hbuf    = (__bf16*)ws; ws += (size_t)kT * kD * 2;
  __bf16* qkvbuf  = (__bf16*)ws; ws += (size_t)kT * kQKV * 2;
  __bf16* attnbuf = (__bf16*)ws; ws += (size_t)kT * kD * 2;
  __bf16* ffbuf   = (__bf16*)ws; ws += (size_t)kT * kM * 2;
  __bf16* WqkvT   = (__bf16*)ws; ws += (size_t)kL * kQKV * kD * 2;
  __bf16* WoutT   = (__bf16*)ws; ws += (size_t)kL * kD * kD * 2;
  __bf16* W1T     = (__bf16*)ws; ws += (size_t)kL * kM * kD * 2;
  __bf16* W2T     = (__bf16*)ws; ws += (size_t)kL * kD * kM * 2;

  hipMemcpyAsync(xbuf, x_in, (size_t)kT * kD * sizeof(float),
                 hipMemcpyDeviceToDevice, stream);

  const dim3 tb(32, 8);
  for (int li = 0; li < kL; li++) {
    transpose_cvt_kernel<<<dim3(kQKV / 32, kD / 32), tb, 0, stream>>>(
        w_qkv + (size_t)li * kD * kQKV, WqkvT + (size_t)li * kQKV * kD, kD, kQKV);
    transpose_cvt_kernel<<<dim3(kD / 32, kD / 32), tb, 0, stream>>>(
        w_out + (size_t)li * kD * kD, WoutT + (size_t)li * kD * kD, kD, kD);
    transpose_cvt_kernel<<<dim3(kM / 32, kD / 32), tb, 0, stream>>>(
        w1 + (size_t)li * kD * kM, W1T + (size_t)li * kM * kD, kD, kM);
    transpose_cvt_kernel<<<dim3(kD / 32, kM / 32), tb, 0, stream>>>(
        w2 + (size_t)li * kM * kD, W2T + (size_t)li * kD * kM, kM, kD);
  }

  for (int li = 0; li < kL; li++) {
    // --- attention block ---
    ln_kernel<false><<<kT, 256, 0, stream>>>(xbuf, ln1_w + (size_t)li * kD,
                                             ln1_b + (size_t)li * kD, hbuf, nullptr);
    gemm_kernel<0><<<dim3(kQKV / 256, kT / 128), 256, 0, stream>>>(
        hbuf, WqkvT + (size_t)li * kQKV * kD, nullptr, nullptr, nullptr, qkvbuf,
        kQKV, kD);
    attn_kernel<<<dim3(kN / 128, kB * kH), 256, 0, stream>>>(qkvbuf, attnbuf);
    gemm_kernel<1><<<dim3(kD / 256, kT / 128), 256, 0, stream>>>(
        attnbuf, WoutT + (size_t)li * kD * kD, nullptr, xbuf, xbuf, nullptr,
        kD, kD);
    // --- feed-forward block ---
    ln_kernel<false><<<kT, 256, 0, stream>>>(xbuf, ln2_w + (size_t)li * kD,
                                             ln2_b + (size_t)li * kD, hbuf, nullptr);
    gemm_kernel<2><<<dim3(kM / 256, kT / 128), 256, 0, stream>>>(
        hbuf, W1T + (size_t)li * kM * kD, b1 + (size_t)li * kM, nullptr, nullptr,
        ffbuf, kM, kD);
    gemm_kernel<3><<<dim3(kD / 256, kT / 128), 256, 0, stream>>>(
        ffbuf, W2T + (size_t)li * kD * kM, b2 + (size_t)li * kD, xbuf, xbuf,
        nullptr, kD, kM);
  }
  ln_kernel<true><<<kT, 256, 0, stream>>>(xbuf, fn_w, fn_b, nullptr, (float*)d_out);
}
